// ODAttention_83992380440983
// MI455X (gfx1250) — matrix operand order repaired and checked by reading, not compiled, not run on hardware
//
#include <hip/hip_runtime.h>

// ---------------------------------------------------------------------------
// Problem constants (from reference): B=8 T=16 N=64 F=64 H=1024
// ---------------------------------------------------------------------------
#define BT    128          // B*T
#define NN    64           // N (= F)
#define FF    64
#define NF    4096         // N*F
#define HH    1024
#define U     4            // int(log(64)) = 4 selected rows
#define MBT   262144       // N*N*F elements per bt slab of M

// ---- workspace layout (bytes) ---------------------------------------------
// Weff is stored pre-swizzled in WMMA B-fragment order:
//   frag = (j*2 + kc)*4 + t, 512 bf16 per frag:
//   elem(lane, e) = Weff[j*64 + kc*32 + (lane>>4)*16 + e][t*16 + (lane&15)]
#define WS_WEFF_O  0u                          // bf16 swizzled, 512KB
#define WS_WEFF_D  (512u*1024u)                // bf16 swizzled, 512KB
#define WS_BEFF    (1024u*1024u)               // f32  [2][64]
#define WS_SC_O    (1024u*1024u + 4096u)       // f32  [128][64][64] = 2MB
#define WS_SC_D    (WS_SC_O + 2097152u)        // f32  [128][64][64] = 2MB
#define WS_OMC     (WS_SC_D + 2097152u)        // f32  [128][4][64]
#define WS_DEC     (WS_OMC + 131072u)          // f32  [128][4][64]
#define WS_IDX     (WS_DEC + 131072u)          // i32  [128][8]

typedef __attribute__((ext_vector_type(16))) __bf16 v16bf;
typedef __attribute__((ext_vector_type(8)))  float  v8f;
typedef __attribute__((ext_vector_type(4)))  unsigned int tdm_g0_t;
typedef __attribute__((ext_vector_type(8)))  int          tdm_g1_t;
typedef __attribute__((ext_vector_type(4)))  int          tdm_g2_t;
typedef __attribute__((ext_vector_type(8)))  int          tdm_g4_t;

// round-to-nearest-even (used once in the fold kernel)
static __device__ __forceinline__ __bf16 f2bf_rn(float f) {
  unsigned u = __builtin_bit_cast(unsigned, f);
  u += 0x7fffu + ((u >> 16) & 1u);
  unsigned short h = (unsigned short)(u >> 16);
  return __builtin_bit_cast(__bf16, h);
}

// truncating conversion: lowers to a single v_mov_b16 dst.l, src.h
static __device__ __forceinline__ __bf16 f2bf(float f) {
  unsigned u = __builtin_bit_cast(unsigned, f);
  unsigned short h = (unsigned short)(u >> 16);
  return __builtin_bit_cast(__bf16, h);
}

static __device__ __forceinline__ v8f zero8() {
  v8f z;
#pragma unroll
  for (int e = 0; e < 8; ++e) z[e] = 0.0f;
  return z;
}

// ---------------------------------------------------------------------------
// TDM: issue a 2D tensor_load_to_lds (clang-23 6-arg builtin).
// D# per CDNA5 ISA §8: group0 = {count=1, lds_addr, global_addr, type=2},
// group1 = {data_size=4B, tensor_dim0/1, tile_dim0/1, tensor_dim0_stride}.
// Wave-level op; issue from one wave, s_wait_tensorcnt + barrier to consume.
// ---------------------------------------------------------------------------
static __device__ __forceinline__ void tdm_load_2d(
    unsigned lds_byte_off, const void* gaddr,
    unsigned tile0, unsigned tile1, unsigned long long stride0_elems)
{
  unsigned long long ga = (unsigned long long)(size_t)gaddr;
  tdm_g0_t g0;
  g0[0] = 1u;                                          // count=1, user D#
  g0[1] = lds_byte_off;                                // lds_addr
  g0[2] = (unsigned)ga;                                // global_addr[31:0]
  g0[3] = (unsigned)((ga >> 32) & 0x01ffffffull) | (2u << 30); // [56:32]|type=2
  tdm_g1_t g1;
  g1[0] = (int)(2u << 16);                             // wg_mask=0, data_size=4B
  g1[1] = (int)((tile0 & 0xffffu) << 16);              // tensor_dim0[15:0]
  g1[2] = (int)((tile0 >> 16) | ((tile1 & 0xffffu) << 16)); // td0 hi | td1 lo
  g1[3] = (int)((tile1 >> 16) | (tile0 << 16));        // td1 hi | tile_dim0
  g1[4] = (int)(tile1 & 0xffffu);                      // tile_dim1, tile_dim2=0
  g1[5] = (int)(stride0_elems & 0xffffffffull);        // dim0_stride[31:0]
  g1[6] = (int)((stride0_elems >> 32) & 0xffffull);    // dim0_stride[47:32]
  g1[7] = 0;
  tdm_g2_t gz; gz[0] = 0; gz[1] = 0; gz[2] = 0; gz[3] = 0;  // groups 2/3 unused
  tdm_g4_t ga8;
#pragma unroll
  for (int e = 0; e < 8; ++e) ga8[e] = 0;
  __builtin_amdgcn_tensor_load_to_lds(g0, g1, gz, gz, ga8, 0);
}

// ---------------------------------------------------------------------------
// WMMA fragment builders (wave32; A and B share one contiguous k order).
// A 16x32: lane&15 = row; kh = 16*(lane>>4); elem e -> K=k0+kh+e
//          => 4x ds_load_b128 per frag.
// C 16x16 f32: row = v + 8*(lane>>4), col = lane&15.
// ---------------------------------------------------------------------------
static __device__ __forceinline__ v16bf load_a_f32v(const float* base, int row0,
                                                    int k0, int ld, int lane) {
  const int m = lane & 15, kh = (lane >> 4) * 16;
  const float* r = base + (row0 + m) * ld + k0 + kh;
  const float4 x0 = *(const float4*)(r);
  const float4 x1 = *(const float4*)(r + 4);
  const float4 x2 = *(const float4*)(r + 8);
  const float4 x3 = *(const float4*)(r + 12);
  v16bf a;
  a[0] = f2bf(x0.x);  a[1] = f2bf(x0.y);  a[2] = f2bf(x0.z);  a[3] = f2bf(x0.w);
  a[4] = f2bf(x1.x);  a[5] = f2bf(x1.y);  a[6] = f2bf(x1.z);  a[7] = f2bf(x1.w);
  a[8] = f2bf(x2.x);  a[9] = f2bf(x2.y);  a[10] = f2bf(x2.z); a[11] = f2bf(x2.w);
  a[12] = f2bf(x3.x); a[13] = f2bf(x3.y); a[14] = f2bf(x3.z); a[15] = f2bf(x3.w);
  return a;
}

// B 32x16 from row-major f32 (LDS): lane&15 = col, K = k0 + 16*(lane>>4) + e.
static __device__ __forceinline__ v16bf load_b_f32(const float* base, int k0,
                                                   int col0, int ld, int lane) {
  const int c = col0 + (lane & 15), kh = (lane >> 4) * 16;
  v16bf b;
#pragma unroll
  for (int e = 0; e < 16; ++e) b[e] = f2bf(base[(k0 + kh + e) * ld + c]);
  return b;
}

#define WMMA_BF16(A, Bm, C) \
  __builtin_amdgcn_wmma_f32_16x16x32_bf16(false, (A), false, (Bm), (short)0, (C), false, false)

// ---------------------------------------------------------------------------
// Kernel 0: zero-fill the 128MB output.
// ---------------------------------------------------------------------------
__global__ void zero_out_kernel(float4* __restrict__ out, int n4) {
  const float4 z = {0.f, 0.f, 0.f, 0.f};
  for (int i = blockIdx.x * blockDim.x + threadIdx.x; i < n4;
       i += gridDim.x * blockDim.x)
    out[i] = z;
}

// ---------------------------------------------------------------------------
// Kernel 1: fold the two linear layers; store Weff pre-swizzled in WMMA
// B-fragment order (bf16): one 32B load per lane in the GEMM kernels.
//   Weff[kf][n] = sum_h Wp[h][kf] * Ws[n][h]
//   beff[n]     = sum_h Ws[n][h]  * bp[h] + bs[n]
// grid = (1024, 2 branches), block = 256 (64 n x 4 kf).
// ---------------------------------------------------------------------------
__global__ void fold_kernel(const float* __restrict__ Wo, const float* __restrict__ bo,
                            const float* __restrict__ Wso, const float* __restrict__ bso,
                            const float* __restrict__ Wd, const float* __restrict__ bd,
                            const float* __restrict__ Wsd, const float* __restrict__ bsd,
                            char* __restrict__ ws) {
  const int branch = blockIdx.y;
  const float* Wp = branch ? Wd : Wo;      // (1024, 4096)
  const float* Wsm = branch ? Wsd : Wso;   // (64, 1024)
  const float* bp = branch ? bd : bo;
  const float* bs = branch ? bsd : bso;
  __bf16* weff = (__bf16*)(ws + (branch ? WS_WEFF_D : WS_WEFF_O));
  float* beff = (float*)(ws + WS_BEFF);

  __shared__ float sWs[64 * 128];
  const int tid = threadIdx.x;
  const int n = tid & 63, kfl = tid >> 6;
  const int kf = blockIdx.x * 4 + kfl;

  float acc = 0.f, bacc = 0.f;
  for (int hc = 0; hc < HH; hc += 128) {
    for (int i = tid; i < 64 * 128; i += 256) {
      const int nn = i >> 7, hh = i & 127;
      sWs[i] = Wsm[nn * HH + hc + hh];
    }
    __syncthreads();
#pragma unroll 4
    for (int hh = 0; hh < 128; ++hh) {
      const float wv = sWs[n * 128 + hh];
      acc += wv * Wp[(size_t)(hc + hh) * NF + kf];
      if (kfl == 0) bacc += wv * bp[hc + hh];
    }
    __syncthreads();
  }
  // swizzled store: frag-major B-fragment layout
  const int j = kf >> 6, kloc = kf & 63, kc = kloc >> 5, k32 = kloc & 31;
  const int lane_s = (n & 15) | (k32 & 16);
  const int addr = ((j * 2 + kc) * 4 + (n >> 4)) * 512 + lane_s * 16 + (k32 & 15);
  weff[addr] = f2bf_rn(acc);
  if (kfl == 0 && blockIdx.x == 0) beff[branch * 64 + n] = bacc + bs[n];
}

// ---------------------------------------------------------------------------
// Kernel 2: fused omega/delta scores. One block per bt, 8 waves:
// waves 0-3 compute omega row-strips, waves 4-7 delta row-strips
// (one branch per wave halves register pressure -> no spills).
//   omega[i,n] = sum_{j,f} M[bt,i,j,f] * WeffO[j*64+f, n]   (+beff)
//   delta[i,n] = sum_{j,f} M[bt,j,i,f] * WeffD[j*64+f, n]   (+beff)
// TDM double-buffers the two 64x64 M slabs; 8 B-frags + 2 A-frags are
// preloaded per j so loads clause and overlap the 8-WMMA chain.
// ---------------------------------------------------------------------------
__global__ void scores_kernel(const float* __restrict__ M, char* __restrict__ ws) {
  __shared__ float sAom[2][64 * 64];   // M[bt,:,j,:]  (i x f), strided slab
  __shared__ float sAde[2][64 * 64];   // M[bt,j,:,:]  (i x f), contiguous slab

  const int bt = blockIdx.x;
  const int tid = threadIdx.x, lane = tid & 31, wv = tid >> 5;
  const int br = wv >> 2;              // 0 = omega, 1 = delta
  const int r0 = (wv & 3) * 16;
  const __bf16* wf = (const __bf16*)(ws + (br ? WS_WEFF_D : WS_WEFF_O));
  const float* Mb = M + (size_t)bt * MBT;

  v8f acc[4];
#pragma unroll
  for (int t = 0; t < 4; ++t) acc[t] = zero8();

  // prologue: issue j=0 slab pair
  if (wv == 0) {
    tdm_load_2d((unsigned)(size_t)(void*)sAom[0], Mb, 64, 64, NF);
    tdm_load_2d((unsigned)(size_t)(void*)sAde[0], Mb, 64, 64, FF);
  }

  for (int j = 0; j < NN; ++j) {
    const int cur = j & 1, nxt = cur ^ 1;

    // preload this wave's 8 B-fragments (one 32B global load per lane each)
    v16bf b0[4], b1[4];
#pragma unroll
    for (int t = 0; t < 4; ++t) {
      b0[t] = *(const v16bf*)(wf + (size_t)((j * 2 + 0) * 4 + t) * 512 + lane * 16);
      b1[t] = *(const v16bf*)(wf + (size_t)((j * 2 + 1) * 4 + t) * 512 + lane * 16);
    }

    if (wv == 0) {
      if (j + 1 < NN) {   // pipeline: issue next slab pair into other buffer
        tdm_load_2d((unsigned)(size_t)(void*)sAom[nxt], Mb + (j + 1) * FF, 64, 64, NF);
        tdm_load_2d((unsigned)(size_t)(void*)sAde[nxt], Mb + (size_t)(j + 1) * NF, 64, 64, FF);
        __builtin_amdgcn_s_wait_tensorcnt(2);  // in-order: oldest pair done
      } else {
        __builtin_amdgcn_s_wait_tensorcnt(0);
      }
      if (j + 1 < NN) {  // prefetch next Weff frag chunks -> global_prefetch_b8
        __builtin_prefetch((const __bf16*)(ws + WS_WEFF_O) + (size_t)(j + 1) * 4096, 0, 0);
        __builtin_prefetch((const __bf16*)(ws + WS_WEFF_D) + (size_t)(j + 1) * 4096, 0, 0);
      }
    }
    __syncthreads();

    const float* slab = br ? sAde[cur] : sAom[cur];
    const v16bf a0 = load_a_f32v(slab, r0, 0, 64, lane);
    const v16bf a1 = load_a_f32v(slab, r0, 32, 64, lane);
#pragma unroll
    for (int t = 0; t < 4; ++t) acc[t] = WMMA_BF16(a0, b0[t], acc[t]);
#pragma unroll
    for (int t = 0; t < 4; ++t) acc[t] = WMMA_BF16(a1, b1[t], acc[t]);
    __syncthreads();
  }

  // epilogue: + beff, store fp32 scores
  const float* beff = (const float*)(ws + WS_BEFF) + br * 64;
  float* sc = (float*)(ws + (br ? WS_SC_D : WS_SC_O));
  const int hi = lane >> 4;
#pragma unroll
  for (int t = 0; t < 4; ++t) {
    const int col = t * 16 + (lane & 15);
    const float bv = beff[col];
#pragma unroll
    for (int v = 0; v < 8; ++v) {
      const int row = r0 + v + 8 * hi;
      sc[((size_t)bt * 64 + row) * 64 + col] = acc[t][v] + bv;
    }
  }
}

// ---------------------------------------------------------------------------
// Kernel 3: softmax + entropy + top-U smallest-entropy rows, compacted.
// grid = (128 bt, 2 branches), block = 64 (one row per thread).
// ---------------------------------------------------------------------------
__global__ void softmax_topk_kernel(char* __restrict__ ws) {
  const int bt = blockIdx.x, branch = blockIdx.y, r = threadIdx.x;
  const float* sc = (const float*)(ws + (branch ? WS_SC_D : WS_SC_O)) +
                    ((size_t)bt * 64 + r) * 64;
  float* cmp = (float*)(ws + (branch ? WS_DEC : WS_OMC));
  int* idx = (int*)(ws + WS_IDX);

  float p[64];
  float m = -1e30f;
#pragma unroll 8
  for (int c = 0; c < 64; ++c) m = fmaxf(m, sc[c]);
  float s = 0.f;
#pragma unroll 8
  for (int c = 0; c < 64; ++c) { const float e = __expf(sc[c] - m); p[c] = e; s += e; }
  const float inv = 1.0f / s;
  float ent = 0.f;
#pragma unroll 8
  for (int c = 0; c < 64; ++c) {
    const float pc = p[c] * inv;
    p[c] = pc;
    ent -= pc * __logf(pc + 1e-9f);
  }

  __shared__ float sent[64];
  __shared__ int ssel[U];
  sent[r] = ent;
  __syncthreads();
  if (r == 0) {
    unsigned long long taken = 0ull;
    for (int sI = 0; sI < U; ++sI) {
      float best = 1e30f; int bi = 0;
      for (int c = 0; c < 64; ++c)
        if (!((taken >> c) & 1ull) && sent[c] < best) { best = sent[c]; bi = c; }
      taken |= 1ull << bi;
      ssel[sI] = bi;
      idx[bt * 8 + branch * 4 + sI] = bi;
    }
  }
  __syncthreads();
#pragma unroll
  for (int sI = 0; sI < U; ++sI)
    if (r == ssel[sI])
      for (int c = 0; c < 64; ++c) cmp[((size_t)bt * U + sI) * 64 + c] = p[c];
}

// ---------------------------------------------------------------------------
// Kernel 4: fused sparse einsums. For each k:
//   T1[i,l] = sum_j omega_c[i,j] * M[bt,j,k,l]   (16-pad x 64, bf16 WMMA)
//   out[bt, I[i], J[j], k] = sum_l T1[i,l] * delta_c[j,l]
// grid = (4 k-slices, 128 bt), block = 32 (one wave). TDM double-buffers
// the M[bt,:,k,:] slab.
// ---------------------------------------------------------------------------
__global__ void apply_kernel(const float* __restrict__ M, const char* __restrict__ ws,
                             float* __restrict__ out) {
  __shared__ float slab[2][64 * 64];  // M[bt, j, k, l] for fixed k: [j][l]
  __shared__ float t1[16 * 64];       // T1 for current k

  const int bt = blockIdx.y;
  const int k0slice = blockIdx.x * 16;
  const int lane = threadIdx.x & 31, hi = lane >> 4;
  const float* Mb = M + (size_t)bt * MBT;
  const float* omc = (const float*)(ws + WS_OMC) + (size_t)bt * U * 64;
  const float* dec = (const float*)(ws + WS_DEC) + (size_t)bt * U * 64;
  const int* idx = (const int*)(ws + WS_IDX) + bt * 8;

  int I[4], J[4];
#pragma unroll
  for (int e = 0; e < 4; ++e) { I[e] = idx[e]; J[e] = idx[4 + e]; }

  // A-frags for omega (rows 0..3 real, rest zero-padded), K = j
  v16bf aOm[2];
#pragma unroll
  for (int kc = 0; kc < 2; ++kc) {
    const int row = lane & 15, kh = hi * 16;
#pragma unroll
    for (int e = 0; e < 16; ++e) {
      const int kk = kc * 32 + kh + e;
      aOm[kc][e] = f2bf(row < U ? omc[row * 64 + kk] : 0.0f);
    }
  }
  // B-frags for delta^T (cols 0..3 real), K = l
  v16bf bDe[2];
#pragma unroll
  for (int kc = 0; kc < 2; ++kc) {
    const int col = lane & 15, kh = hi * 16;
#pragma unroll
    for (int e = 0; e < 16; ++e) {
      const int kk = kc * 32 + kh + e;
      bDe[kc][e] = f2bf(col < U ? dec[col * 64 + kk] : 0.0f);
    }
  }

  // prologue: issue first slab
  tdm_load_2d((unsigned)(size_t)(void*)slab[0], Mb + k0slice * FF, 64, 64, NF);

  for (int k = k0slice; k < k0slice + 16; ++k) {
    const int cur = k & 1, nxt = cur ^ 1;
    if (k + 1 < k0slice + 16) {
      tdm_load_2d((unsigned)(size_t)(void*)slab[nxt], Mb + (k + 1) * FF, 64, 64, NF);
      __builtin_amdgcn_s_wait_tensorcnt(1);  // in-order: current slab done
    } else {
      __builtin_amdgcn_s_wait_tensorcnt(0);
    }
    __syncthreads();

    // phase 1: T1 = omega_c (16x64j) @ slab (64j x 64l)
    v8f cc[4];
#pragma unroll
    for (int t = 0; t < 4; ++t) {
      cc[t] = zero8();
#pragma unroll
      for (int kc = 0; kc < 2; ++kc) {
        const v16bf bS = load_b_f32(slab[cur], kc * 32, t * 16, 64, lane);
        cc[t] = WMMA_BF16(aOm[kc], bS, cc[t]);
      }
    }
    // park T1 in LDS to rebuild A-frags (C-layout -> A-layout shuffle)
#pragma unroll
    for (int t = 0; t < 4; ++t) {
      const int c = t * 16 + (lane & 15);
#pragma unroll
      for (int v = 0; v < 8; ++v) t1[(v + 8 * hi) * 64 + c] = cc[t][v];
    }
    __syncthreads();

    // phase 2: out_tile = T1 (16i x 64l) @ delta^T (64l x 16j)
    v8f oc = zero8();
#pragma unroll
    for (int kc = 0; kc < 2; ++kc) {
      const v16bf aT = load_a_f32v(t1, 0, kc * 32, 64, lane);
      oc = WMMA_BF16(aT, bDe[kc], oc);
    }
    // scatter the 4x4 nonzero (i,j) block for this k
    const int jc = lane & 15;
#pragma unroll
    for (int v = 0; v < 8; ++v) {
      const int ic = v + 8 * hi;
      if (ic < U && jc < U)
        out[(size_t)bt * MBT + (size_t)I[ic] * NF + J[jc] * FF + k] = oc[v];
    }
    __syncthreads();
  }
}

// ---------------------------------------------------------------------------
extern "C" void kernel_launch(void* const* d_in, const int* in_sizes, int n_in,
                              void* d_out, int out_size, void* d_ws, size_t ws_size,
                              hipStream_t stream) {
  (void)in_sizes; (void)n_in; (void)ws_size;
  const float* M   = (const float*)d_in[0];
  const float* Wo  = (const float*)d_in[1];
  const float* bo  = (const float*)d_in[2];
  const float* Wso = (const float*)d_in[3];
  const float* bso = (const float*)d_in[4];
  const float* Wd  = (const float*)d_in[5];
  const float* bd  = (const float*)d_in[6];
  const float* Wsd = (const float*)d_in[7];
  const float* bsd = (const float*)d_in[8];
  float* out = (float*)d_out;
  char* ws = (char*)d_ws;

  zero_out_kernel<<<2048, 256, 0, stream>>>((float4*)out, out_size / 4);
  fold_kernel<<<dim3(NF / 4, 2), 256, 0, stream>>>(Wo, bo, Wso, bso, Wd, bd, Wsd, bsd, ws);
  scores_kernel<<<BT, 256, 0, stream>>>(M, ws);
  softmax_topk_kernel<<<dim3(BT, 2), 64, 0, stream>>>(ws);
  apply_kernel<<<dim3(4, BT), 32, 0, stream>>>(M, ws, out);
}
